// MultiHead_BlockAttention_42949673148
// MI455X (gfx1250) — matrix operand order repaired and checked by reading, not compiled, not run on hardware
//
#include <hip/hip_runtime.h>
#include <hip/hip_bf16.h>

#define HN 8
#define BB 16
#define SS 512
#define FF 512
#define CC 4096   // S*H
#define DD 512    // head dim

typedef __attribute__((ext_vector_type(16))) _Float16 v16h;
typedef __attribute__((ext_vector_type(8)))  _Float16 v8h;
typedef __attribute__((ext_vector_type(8)))  float    v8f;

union ABu { v16h v; v8h h[2]; };

// A fragment (16x32 f16, M x K), src row-major with leading dim ld (halfs).
// Lane layout (ISA 7.12.2): row = lane%16; elems 0..7 = K=kb..kb+7,
// elems 8..15 = K=16+kb..16+kb+7, kb = (lane>=16) ? 8 : 0.
__device__ __forceinline__ v16h load_a_rm(const _Float16* base, int ld, int k0, int lane) {
  const int row = lane & 15;
  const int kb  = (lane >> 4) << 3;
  const _Float16* p = base + (size_t)row * ld + k0 + kb;
  ABu u;
  u.h[0] = *(const v8h*)(p);
  u.h[1] = *(const v8h*)(p + 16);
  return u.v;
}

// B fragment (32x16 f16, K x N), src stored N-major (bT[n][k], leading dim ld).
// Lane layout: col n = lane%16; elems 0..7 = K=kb..kb+7, elems 8..15 = K=16+kb..16+kb+7, kb = (lane>=16) ? 8 : 0.
__device__ __forceinline__ v16h load_b_nm(const _Float16* bT, int ld, int k0, int lane) {
  const int n  = lane & 15;
  const int kb = (lane >> 4) << 3;
  const _Float16* p = bT + (size_t)n * ld + k0 + kb;
  ABu u;
  u.h[0] = *(const v8h*)(p);
  u.h[1] = *(const v8h*)(p + 16);
  return u.v;
}

__device__ __forceinline__ v8f wmma_f16(v16h a, v16h b, v8f c) {
  return __builtin_amdgcn_wmma_f32_16x16x32_f16(false, a, false, b, (short)0, c, false, false);
}

// ---- f32 -> f16 convert of x ----
__global__ void cvt_x(const float* __restrict__ x, _Float16* __restrict__ x16, int n) {
  int i = blockIdx.x * blockDim.x + threadIdx.x;
  if (i < n) x16[i] = (_Float16)x[i];
}

// ---- transpose+convert all three weights into wT[3][CC][FF] (N-major) ----
__global__ void cvt_wT(const float* __restrict__ wq, const float* __restrict__ wk,
                       const float* __restrict__ wv, _Float16* __restrict__ wT) {
  int i = blockIdx.x * blockDim.x + threadIdx.x;
  const int per = CC * FF;
  if (i >= 3 * per) return;
  int mat = i / per;
  int r   = i - mat * per;
  int c   = r / FF;
  int f   = r - c * FF;
  const float* w = (mat == 0) ? wq : (mat == 1) ? wk : wv;
  wT[i] = (_Float16)w[(size_t)f * CC + c];
}

// ---- K and V projection GEMMs, with LDS-staged coalesced stores ----
// K stored row-major per head: k16[h][b][s][d]  (wave tile 16 s x 64 d)
// V stored d-major per head:  vT16[h][b][d][s]  (wave tile 64 s x 16 d, transposed in LDS)
#define SPITCH 72   // halfs; 144B rows keep v8h reads 16B-aligned, conflict-free transpose
__global__ void __launch_bounds__(256)
proj_kv(const _Float16* __restrict__ x16, const _Float16* __restrict__ wT,
        _Float16* __restrict__ k16, _Float16* __restrict__ vT16) {
  __shared__ __align__(16) _Float16 stage[8][16][SPITCH];

  const int lane = threadIdx.x & 31;
  const int wv_  = threadIdx.x >> 5;
  const int wave = blockIdx.x * 8 + wv_;
  const int mat  = wave >> 15;            // 0 = K, 1 = V (uniform per block)
  const int rem  = wave & 32767;
  const int mbase = (lane >> 4) << 3;
  const int n     = lane & 15;

  int m0, c0;                              // tile origin: rows in [0,B*S), cols in [0,CC)
  if (mat == 0) { m0 = (rem >> 6) << 4;  c0 = (rem & 63) << 6; }   // 16 x 64
  else          { m0 = (rem >> 8) << 6;  c0 = (rem & 255) << 4; }  // 64 x 16

  const _Float16* A  = x16 + (size_t)m0 * FF;
  const _Float16* Bt = wT + (size_t)(mat + 1) * CC * FF;           // wk or wv, N-major

  v8f acc[4] = {};
  if (mat == 0) {
    for (int k0 = 0; k0 < FF; k0 += 32) {
      v16h a = load_a_rm(A, FF, k0, lane);
#pragma unroll
      for (int t = 0; t < 4; ++t) {
        v16h bf = load_b_nm(Bt + (size_t)(c0 + t * 16) * FF, FF, k0, lane);
        acc[t] = wmma_f16(a, bf, acc[t]);
      }
    }
    // stage[row = s-local][col = d-local]
#pragma unroll
    for (int t = 0; t < 4; ++t)
#pragma unroll
      for (int r = 0; r < 8; ++r)
        stage[wv_][mbase + r][t * 16 + n] = (_Float16)acc[t][r];
  } else {
    for (int k0 = 0; k0 < FF; k0 += 32) {
      v16h bf = load_b_nm(Bt + (size_t)c0 * FF, FF, k0, lane);
#pragma unroll
      for (int t = 0; t < 4; ++t) {
        v16h a = load_a_rm(A + (size_t)(t * 16) * FF, FF, k0, lane);
        acc[t] = wmma_f16(a, bf, acc[t]);
      }
    }
    // transpose into stage[row = d-local][col = s-local]
#pragma unroll
    for (int t = 0; t < 4; ++t)
#pragma unroll
      for (int r = 0; r < 8; ++r)
        stage[wv_][n][t * 16 + mbase + r] = (_Float16)acc[t][r];
  }
  __syncthreads();

  // Coalesced writeout: 16 rows x 64 halfs (128B contiguous per row).
  // 8 lanes per row, 16B (v8h) per lane, 4 row-groups.
  {
    const int b  = m0 >> 9;
    const int h  = c0 >> 9;
    const size_t hb = (size_t)(h * BB + b) * SS * DD;
    const int seg = lane & 7;            // 8 x 8 halfs = 64 halfs per row
#pragma unroll
    for (int g = 0; g < 4; ++g) {
      const int row = g * 4 + (lane >> 3);
      v8h val = *(const v8h*)&stage[wv_][row][seg * 8];
      _Float16* dst;
      if (mat == 0) {
        const int s = (m0 & 511) + row;          // output row = s, cols = d
        dst = k16 + hb + (size_t)s * DD + (c0 & 511) + seg * 8;
      } else {
        const int d = (c0 & 511) + row;          // output row = d, cols = s
        dst = vT16 + hb + (size_t)d * SS + (m0 & 511) + seg * 8;
      }
      *(v8h*)dst = val;
    }
  }
}

// ---- fused attention: Q on the fly, logits, softmax, P@V ----
__global__ void __launch_bounds__(256)
attn(const _Float16* __restrict__ x16, const _Float16* __restrict__ wT,
     const _Float16* __restrict__ k16, const _Float16* __restrict__ vT16,
     float* __restrict__ out) {
  __shared__ __align__(16) _Float16 qp[16][512];    // Qt (phases 0/A), P16 (phases B/C)
  __shared__ float Llog[16][512];
  __shared__ float red[16][16];
  __shared__ float rowinv[16];

  const int tid  = threadIdx.x;
  const int lane = tid & 31;
  const int w    = tid >> 5;                 // wave 0..7
  const int mbase = (lane >> 4) << 3;
  const int n     = lane & 15;

  const int blk   = blockIdx.x;
  const int h     = blk / (BB * 32);
  int rem         = blk - h * (BB * 32);
  const int b     = rem >> 5;
  const int itile = rem & 31;
  const int i0    = itile << 4;

  const size_t hb = (size_t)(h * BB + b) * SS * DD;
  const _Float16* Kp  = k16 + hb;                               // [j][d] row-major
  const _Float16* Vt  = vT16 + hb;                              // [d][j] d-major
  const _Float16* Xa  = x16 + ((size_t)b * SS + i0) * FF;       // 16 x 512 rows
  const _Float16* WqT = wT + (size_t)h * DD * FF;               // wq cols h*512.., N-major

  // Phase 0: Q tile (16x512) = Xa @ Wq_head, into LDS (f16)
  {
    v8f acc[4] = {};
    const int d0 = w * 64;
    for (int k0 = 0; k0 < FF; k0 += 32) {
      v16h a = load_a_rm(Xa, FF, k0, lane);
#pragma unroll
      for (int t = 0; t < 4; ++t) {
        v16h bf = load_b_nm(WqT + (size_t)(d0 + t * 16) * FF, FF, k0, lane);
        acc[t] = wmma_f16(a, bf, acc[t]);
      }
    }
#pragma unroll
    for (int t = 0; t < 4; ++t)
#pragma unroll
      for (int r = 0; r < 8; ++r)
        qp[mbase + r][d0 + t * 16 + n] = (_Float16)acc[t][r];
  }
  __syncthreads();

  // Phase A: logits (16x512) = Qt @ K^T, f32 into LDS; wave w owns j in [w*64, w*64+64)
  {
    v8f acc[4] = {};
    const int j0w = w * 64;
    for (int k0 = 0; k0 < DD; k0 += 32) {
      v16h a = load_a_rm(&qp[0][0], 512, k0, lane);
#pragma unroll
      for (int t = 0; t < 4; ++t) {
        v16h bf = load_b_nm(Kp + (size_t)(j0w + t * 16) * DD, DD, k0, lane);
        acc[t] = wmma_f16(a, bf, acc[t]);
      }
    }
#pragma unroll
    for (int t = 0; t < 4; ++t)
#pragma unroll
      for (int r = 0; r < 8; ++r)
        Llog[mbase + r][j0w + t * 16 + n] = acc[t][r];
  }
  __syncthreads();

  // Phase B: row softmax (16 rows x 512), 16 threads per row; P -> qp (f16)
  {
    const int row = tid >> 4;
    const int seg = tid & 15;
    float mx = -3.4e38f;
#pragma unroll
    for (int c = 0; c < 32; ++c) mx = fmaxf(mx, Llog[row][seg * 32 + c]);
    red[row][seg] = mx;
    __syncthreads();
    float m = -3.4e38f;
#pragma unroll
    for (int t = 0; t < 16; ++t) m = fmaxf(m, red[row][t]);
    __syncthreads();
    float s = 0.f;
    for (int c = 0; c < 32; ++c) {
      int col = seg * 32 + c;
      float e = __expf(Llog[row][col] - m);
      qp[row][col] = (_Float16)e;
      s += e;
    }
    red[row][seg] = s;
    __syncthreads();
    if (tid < 16) {
      float t = 0.f;
#pragma unroll
      for (int q2 = 0; q2 < 16; ++q2) t += red[tid][q2];
      rowinv[tid] = 1.0f / t;
    }
    __syncthreads();
  }

  // Phase C: out tile = (P @ V) * (1/rowsum); wave w owns d in [w*64, w*64+64)
  {
    v8f acc[4] = {};
    const int d0 = w * 64;
    for (int j0 = 0; j0 < SS; j0 += 32) {
      v16h a = load_a_rm(&qp[0][0], 512, j0, lane);
#pragma unroll
      for (int t = 0; t < 4; ++t) {
        v16h bf = load_b_nm(Vt + (size_t)(d0 + t * 16) * SS, SS, j0, lane);
        acc[t] = wmma_f16(a, bf, acc[t]);
      }
    }
#pragma unroll
    for (int t = 0; t < 4; ++t) {
#pragma unroll
      for (int r = 0; r < 8; ++r) {
        int mrow = mbase + r;
        int s = i0 + mrow;
        int c = h * DD + d0 + t * 16 + n;
        out[((size_t)b * SS + s) * CC + c] = acc[t][r] * rowinv[mrow];
      }
    }
  }
}

extern "C" void kernel_launch(void* const* d_in, const int* in_sizes, int n_in,
                              void* d_out, int out_size, void* d_ws, size_t ws_size,
                              hipStream_t stream) {
  (void)in_sizes; (void)n_in; (void)out_size; (void)ws_size;
  const float* x  = (const float*)d_in[0];
  const float* wq = (const float*)d_in[1];
  const float* wk = (const float*)d_in[2];
  const float* wv = (const float*)d_in[3];
  float* out = (float*)d_out;

  char* ws = (char*)d_ws;
  size_t off = 0;
  _Float16* x16  = (_Float16*)(ws + off); off += (size_t)BB * SS * FF * 2;       // 8.4 MB
  _Float16* wT   = (_Float16*)(ws + off); off += (size_t)3 * CC * FF * 2;        // 12.6 MB
  _Float16* k16  = (_Float16*)(ws + off); off += (size_t)HN * BB * SS * DD * 2;  // 67 MB
  _Float16* vT16 = (_Float16*)(ws + off); off += (size_t)HN * BB * SS * DD * 2;  // 67 MB

  {
    int nel = BB * SS * FF;
    cvt_x<<<(nel + 255) / 256, 256, 0, stream>>>(x, x16, nel);
  }
  {
    int nel = 3 * CC * FF;
    cvt_wT<<<(nel + 255) / 256, 256, 0, stream>>>(wq, wk, wv, wT);
  }
  {
    int waves = 2 * 32768;                 // 32768 K-units + 32768 V-units
    proj_kv<<<waves / 8, 256, 0, stream>>>(x16, wT, k16, vT16);
  }
  attn<<<HN * BB * 32, 256, 0, stream>>>(x16, wT, k16, vT16, out);
}
